// MAN_1211180777572
// MI455X (gfx1250) — compile-verified, matrix operand order checked by reading, not run on hardware
//
#include <hip/hip_runtime.h>
#include <hip/hip_bf16.h>

// ---------------------------------------------------------------------------
// Problem constants (match reference)
// ---------------------------------------------------------------------------
constexpr int NT = 1024;   // tweets
constexpr int L  = 64;     // tokens per tweet
constexpr int E  = 512;    // embedding dim
constexpr int H  = 256;    // LSTM hidden per direction
constexpr int G4 = 4 * H;  // 1024 gate columns
constexpr int XS = E + 8;  // LDS x-tile row stride (pad keeps 16B alignment: 520*2=1040B)

typedef __attribute__((ext_vector_type(16))) _Float16 v16h;
typedef __attribute__((ext_vector_type(8)))  _Float16 v8h;
typedef __attribute__((ext_vector_type(4)))  _Float16 v4h;
typedef __attribute__((ext_vector_type(8)))  float    v8f;

// ---------------------------------------------------------------------------
// WMMA fragment loaders (16-bit A 16x32 layout, CDNA5 ISA 7.12.2):
//   lanes 0-15  : row = lane,    K = k0+0..7  and k0+16..23
//   lanes 16-31 : row = lane-16, K = k0+8..15 and k0+24..31
// B (32x16, = W^T chunk) mirrors this with "row" -> output column, which for
// row-major W[ncols][K] is again a contiguous 16B read along K.
// ---------------------------------------------------------------------------
__device__ inline v16h load_frag16(const _Float16* __restrict__ base,
                                   long long row_stride, int lane) {
  const int r    = lane & 15;
  const int koff = (lane < 16) ? 0 : 8;
  const _Float16* p = base + (long long)r * row_stride + koff;
  v8h lo = *(const v8h*)(p);
  v8h hi = *(const v8h*)(p + 16);
  return __builtin_shufflevector(lo, hi, 0, 1, 2, 3, 4, 5, 6, 7,
                                         8, 9, 10, 11, 12, 13, 14, 15);
}

__device__ inline v16h load_frag16_lds(const _Float16* base,
                                       int row_stride, int lane) {
  const int r    = lane & 15;
  const int koff = (lane < 16) ? 0 : 8;
  const _Float16* p = base + r * row_stride + koff;
  v8h lo = *(const v8h*)(p);
  v8h hi = *(const v8h*)(p + 16);
  return __builtin_shufflevector(lo, hi, 0, 1, 2, 3, 4, 5, 6, 7,
                                         8, 9, 10, 11, 12, 13, 14, 15);
}

__device__ inline v8f wmma_f16(v16h a, v16h b, v8f c) {
  return __builtin_amdgcn_wmma_f32_16x16x32_f16(
      /*neg_a=*/false, a, /*neg_b=*/false, b,
      /*c_mod=*/(short)0, c, /*reuse_a=*/false, /*reuse_b=*/false);
}

__device__ inline float sigmoidf_(float x) { return 1.f / (1.f + __expf(-x)); }

// ---------------------------------------------------------------------------
// Async staging of one 16 x E x_t tile into LDS via the gfx1250 TDM-adjacent
// async path (ASYNCcnt-tracked, per-lane 16B transfers).
// 16 rows * 64 chunks = 1024 chunks / 128 threads = 8 issues per thread.
// ---------------------------------------------------------------------------
__device__ inline void issue_x_tile_async(const _Float16* __restrict__ x,
                                          int m0, int tc, _Float16* dstbase,
                                          int tid) {
#pragma unroll
  for (int i = 0; i < 8; ++i) {
    const int c   = tid + i * 128;
    const int row = c >> 6;          // 0..15
    const int cc  = c & 63;          // 16B chunk within row
    const _Float16* g =
        x + ((long long)(m0 + row) * L + tc) * E + cc * 8;
    const unsigned lds =
        (unsigned)(unsigned long long)(dstbase + row * XS + cc * 8);
    asm volatile("global_load_async_to_lds_b128 %0, %1, off"
                 :: "v"(lds), "v"(g) : "memory");
  }
}

__device__ inline void wait_async_all() {
  asm volatile("s_wait_asynccnt 0" ::: "memory");
}

// ---------------------------------------------------------------------------
// K1: embedding gather + f32 -> f16 convert.  x[(n*L+l)][E]
// One float4 load + one v4h store per thread (128 threads x 4 = 512 = E).
// ---------------------------------------------------------------------------
__global__ __launch_bounds__(128)
void k_gather_embed(const int* __restrict__ tokens,
                    const float* __restrict__ embed,
                    _Float16* __restrict__ x) {
  const long long row = blockIdx.x;            // n*L + l
  const int tok = tokens[row];
  const float4 v =
      ((const float4*)(embed + (long long)tok * E))[threadIdx.x];
  const v4h h = { (_Float16)v.x, (_Float16)v.y, (_Float16)v.z, (_Float16)v.w };
  *(v4h*)(x + row * E + threadIdx.x * 4) = h;
}

// K2: generic f32 -> f16 (weights)
__global__ __launch_bounds__(256)
void k_f32_to_f16(const float* __restrict__ src, _Float16* __restrict__ dst,
                  int n) {
  int i = blockIdx.x * blockDim.x + threadIdx.x;
  if (i < n) dst[i] = (_Float16)src[i];
}

// ---------------------------------------------------------------------------
// K3: fused LSTM direction.  Block = 128 threads = 4 waves, 16 tweets/block.
// Wave w owns gate w (i,f,g,o) = columns [w*H, (w+1)*H).
// Per timestep: G = x_t @ Wih^T + h @ Whh^T + b  via WMMA, gates through LDS.
// x_t tiles are double-buffered in LDS via async-to-LDS, prefetching t+1
// while computing t.  h carried as f16 in LDS across the 64 steps.
// ---------------------------------------------------------------------------
__global__ __launch_bounds__(128)
void k_lstm_dir(const _Float16* __restrict__ x,    // [NT, L, E] f16
                const _Float16* __restrict__ Wih,  // [4H, E]   f16
                const _Float16* __restrict__ Whh,  // [4H, H]   f16
                const float* __restrict__ bih,     // [4H]
                const float* __restrict__ bhh,     // [4H]
                float* __restrict__ h_out,         // [NT, H]
                int reverse) {
  __shared__ __align__(16) _Float16 x_sh[2][16][XS]; // double-buffered x tile
  __shared__ __align__(16) _Float16 h_sh[16][H + 8];
  __shared__ float    c_sh[16][H];
  __shared__ float    g_sh[4][16][H];      // activated gates i,f,g,o

  const int tid  = threadIdx.x;
  const int wave = tid >> 5;
  const int lane = tid & 31;
  const int m0   = blockIdx.x * 16;        // tweet-row base
  const int g0   = wave * H;               // this wave's gate-column base

  for (int idx = tid; idx < 16 * H; idx += 128) {
    const int m = idx >> 8, n = idx & (H - 1);
    h_sh[m][n] = (_Float16)0.f;
    c_sh[m][n] = 0.f;
  }
  // prologue: stage x tile for t = 0
  issue_x_tile_async(x, m0, reverse ? (L - 1) : 0, &x_sh[0][0][0], tid);
  wait_async_all();
  __syncthreads();

  for (int t = 0; t < L; ++t) {
    const int cur = t & 1;

    // prefetch next timestep's x tile into the other buffer (overlaps WMMA)
    if (t + 1 < L) {
      const int tn = reverse ? (L - 2 - t) : (t + 1);
      issue_x_tile_async(x, m0, tn, &x_sh[cur ^ 1][0][0], tid);
    }

    v8f acc[16];
    const v8f vzero = {};
#pragma unroll
    for (int j = 0; j < 16; ++j) acc[j] = vzero;

    // ---- x_t @ Wih^T : K = E = 512 (16 slices of 32), A from LDS -------
    for (int kt = 0; kt < E / 32; ++kt) {
      const v16h a = load_frag16_lds(&x_sh[cur][0][kt * 32], XS, lane);
      __builtin_prefetch(Wih + (long long)g0 * E + ((kt + 1) & 15) * 32, 0, 3);
#pragma unroll
      for (int j = 0; j < 16; ++j) {
        const _Float16* wb = Wih + (long long)(g0 + j * 16) * E + kt * 32;
        const v16h b = load_frag16(wb, E, lane);
        acc[j] = wmma_f16(a, b, acc[j]);
      }
    }

    // ---- h @ Whh^T : K = H = 256 (8 slices of 32), A from LDS ----------
    for (int kt = 0; kt < H / 32; ++kt) {
      const v16h a = load_frag16_lds(&h_sh[0][kt * 32], H + 8, lane);
#pragma unroll
      for (int j = 0; j < 16; ++j) {
        const _Float16* wb = Whh + (long long)(g0 + j * 16) * H + kt * 32;
        const v16h b = load_frag16(wb, H, lane);
        acc[j] = wmma_f16(a, b, acc[j]);
      }
    }

    // ---- bias + activation, stash to LDS -------------------------------
    // C/D layout: VGPR r -> row (lane>=16 ? 8+r : r), col = lane&15
#pragma unroll
    for (int j = 0; j < 16; ++j) {
      const int col   = j * 16 + (lane & 15);
      const int rbase = (lane >> 4) * 8;
      const float bsum = bih[g0 + col] + bhh[g0 + col];
#pragma unroll
      for (int r = 0; r < 8; ++r) {
        float v = acc[j][r] + bsum;
        v = (wave == 2) ? tanhf(v) : sigmoidf_(v);   // gate g uses tanh
        g_sh[wave][rbase + r][col] = v;
      }
    }
    __syncthreads();

    // ---- c, h update ----------------------------------------------------
    for (int idx = tid; idx < 16 * H; idx += 128) {
      const int m = idx >> 8, n = idx & (H - 1);
      const float iv = g_sh[0][m][n];
      const float fv = g_sh[1][m][n];
      const float gv = g_sh[2][m][n];
      const float ov = g_sh[3][m][n];
      const float c  = fv * c_sh[m][n] + iv * gv;
      c_sh[m][n] = c;
      h_sh[m][n] = (_Float16)(ov * tanhf(c));
    }
    // next iteration reads the freshly staged x tile: drain ASYNCcnt first
    wait_async_all();
    __syncthreads();
  }

  for (int idx = tid; idx < 16 * H; idx += 128) {
    const int m = idx >> 8, n = idx & (H - 1);
    h_out[(long long)(m0 + m) * H + n] = (float)h_sh[m][n];
  }
}

// ---------------------------------------------------------------------------
// K4: image mean over 49 positions -> f16
// ---------------------------------------------------------------------------
__global__ __launch_bounds__(128)
void k_image_mean(const float* __restrict__ images, _Float16* __restrict__ out) {
  const int n = blockIdx.x;
  for (int e = threadIdx.x; e < 512; e += blockDim.x) {
    float s = 0.f;
    for (int k = 0; k < 49; ++k)
      s += images[((long long)n * 49 + k) * 512 + e];
    out[(long long)n * 512 + e] = (_Float16)(s * (1.f / 49.f));
  }
}

// ---------------------------------------------------------------------------
// K5: WMMA GEMM  C[M,Nc] = A[M,K] @ Bw[Nc,K]^T + bias   (image projection)
// grid (M/16, Nc/64); 4 waves/block, wave w -> col tile w
// ---------------------------------------------------------------------------
__global__ __launch_bounds__(128)
void k_gemm_bias(const _Float16* __restrict__ A, const _Float16* __restrict__ Bw,
                 const float* __restrict__ bias, float* __restrict__ C,
                 int M, int Nc, int K) {
  const int wave = threadIdx.x >> 5, lane = threadIdx.x & 31;
  const int m0 = blockIdx.x * 16;
  const int n0 = blockIdx.y * 64 + wave * 16;
  v8f acc = {};
  for (int k0 = 0; k0 < K; k0 += 32) {
    const v16h a = load_frag16(A + (long long)m0 * K + k0, K, lane);
    const v16h b = load_frag16(Bw + (long long)n0 * K + k0, K, lane);
    acc = wmma_f16(a, b, acc);
  }
  const int col   = n0 + (lane & 15);
  const int rbase = m0 + (lane >> 4) * 8;
  const float bv = bias[col];
#pragma unroll
  for (int r = 0; r < 8; ++r)
    C[(long long)(rbase + r) * Nc + col] = acc[r] + bv;
}

// ---------------------------------------------------------------------------
// K6: per-tweet 2-way attention fusion
// ---------------------------------------------------------------------------
__global__ __launch_bounds__(128)
void k_fusion(const float* __restrict__ h_f, const float* __restrict__ h_b,
              const float* __restrict__ image_h,
              const float* __restrict__ W_T, const float* __restrict__ W_I,
              float* __restrict__ feature) {
  const int n = blockIdx.x;
  __shared__ float rt[128], ri[128];
  float ts = 0.f, is = 0.f;
  for (int e = threadIdx.x; e < 512; e += 128) {
    const float th = (e < 256) ? h_f[(long long)n * 256 + e]
                               : h_b[(long long)n * 256 + (e - 256)];
    ts += th * W_T[e];
    is += image_h[(long long)n * 512 + e] * W_I[e];
  }
  rt[threadIdx.x] = ts;
  ri[threadIdx.x] = is;
  __syncthreads();
  for (int s = 64; s > 0; s >>= 1) {
    if (threadIdx.x < s) {
      rt[threadIdx.x] += rt[threadIdx.x + s];
      ri[threadIdx.x] += ri[threadIdx.x + s];
    }
    __syncthreads();
  }
  const float tsum = rt[0], isum = ri[0];
  const float mx = fmaxf(tsum, isum);
  const float e0 = __expf(tsum - mx), e1 = __expf(isum - mx);
  const float inv = 1.f / (e0 + e1);
  const float a0 = e0 * inv, a1 = e1 * inv;
  for (int e = threadIdx.x; e < 512; e += 128) {
    const float th = (e < 256) ? h_f[(long long)n * 256 + e]
                               : h_b[(long long)n * 256 + (e - 256)];
    feature[(long long)n * 512 + e] = a0 * th + a1 * image_h[(long long)n * 512 + e];
  }
}

// K7: column mean over 1024 tweets
__global__ __launch_bounds__(128)
void k_col_mean(const float* __restrict__ feature, float* __restrict__ mf) {
  const int c = blockIdx.x * blockDim.x + threadIdx.x;
  if (c >= 512) return;
  float s = 0.f;
  for (int n = 0; n < NT; ++n) s += feature[(long long)n * 512 + c];
  mf[c] = s * (1.f / NT);
}

// K8: classifier MLP (single block)
__global__ __launch_bounds__(256)
void k_classifier(const float* __restrict__ mf,
                  const float* __restrict__ Wc1, const float* __restrict__ bc1,
                  const float* __restrict__ Wc2, const float* __restrict__ bc2,
                  float* __restrict__ out) {
  __shared__ float mfs[512];
  __shared__ float hdd[512];
  __shared__ float red[256];
  for (int i = threadIdx.x; i < 512; i += 256) mfs[i] = mf[i];
  __syncthreads();
  for (int o = threadIdx.x; o < 512; o += 256) {
    float s = bc1[o];
    for (int k = 0; k < 512; ++k) s += mfs[k] * Wc1[(long long)o * 512 + k];
    hdd[o] = fmaxf(s, 0.f);
  }
  __syncthreads();
  for (int o = 0; o < 2; ++o) {
    float s = 0.f;
    for (int k = threadIdx.x; k < 512; k += 256)
      s += hdd[k] * Wc2[(long long)o * 512 + k];
    red[threadIdx.x] = s;
    __syncthreads();
    for (int st = 128; st > 0; st >>= 1) {
      if (threadIdx.x < st) red[threadIdx.x] += red[threadIdx.x + st];
      __syncthreads();
    }
    if (threadIdx.x == 0) out[o] = red[0] + bc2[o];
    __syncthreads();
  }
}

// ---------------------------------------------------------------------------
// Host-side orchestration
// ---------------------------------------------------------------------------
extern "C" void kernel_launch(void* const* d_in, const int* in_sizes, int n_in,
                              void* d_out, int out_size, void* d_ws, size_t ws_size,
                              hipStream_t stream) {
  const int*   tokens = (const int*)  d_in[0];
  const float* images = (const float*)d_in[1];
  const float* embed  = (const float*)d_in[2];
  const float* Wih_f  = (const float*)d_in[3];
  const float* Whh_f  = (const float*)d_in[4];
  const float* bih_f  = (const float*)d_in[5];
  const float* bhh_f  = (const float*)d_in[6];
  const float* Wih_b  = (const float*)d_in[7];
  const float* Whh_b  = (const float*)d_in[8];
  const float* bih_b  = (const float*)d_in[9];
  const float* bhh_b  = (const float*)d_in[10];
  const float* Wimg   = (const float*)d_in[11];
  const float* bimg   = (const float*)d_in[12];
  const float* W_T    = (const float*)d_in[13];
  const float* W_I    = (const float*)d_in[14];
  const float* Wc1    = (const float*)d_in[15];
  const float* bc1    = (const float*)d_in[16];
  const float* Wc2    = (const float*)d_in[17];
  const float* bc2    = (const float*)d_in[18];
  float* out = (float*)d_out;

  // ---- carve workspace -------------------------------------------------
  char* p = (char*)d_ws;
  auto carve = [&](size_t bytes) -> void* {
    void* r = (void*)p;
    p += (bytes + 255) & ~(size_t)255;
    return r;
  };
  _Float16* x16     = (_Float16*)carve((size_t)NT * L * E * 2);   // 64 MB
  _Float16* wihf16  = (_Float16*)carve((size_t)G4 * E * 2);
  _Float16* whhf16  = (_Float16*)carve((size_t)G4 * H * 2);
  _Float16* wihb16  = (_Float16*)carve((size_t)G4 * E * 2);
  _Float16* whhb16  = (_Float16*)carve((size_t)G4 * H * 2);
  _Float16* wimg16  = (_Float16*)carve((size_t)512 * 512 * 2);
  _Float16* imean16 = (_Float16*)carve((size_t)NT * 512 * 2);
  float* h_f        = (float*)carve((size_t)NT * H * 4);
  float* h_b        = (float*)carve((size_t)NT * H * 4);
  float* image_h    = (float*)carve((size_t)NT * 512 * 4);
  float* feature    = (float*)carve((size_t)NT * 512 * 4);
  float* mf         = (float*)carve((size_t)512 * 4);

  // ---- weight conversions to f16 --------------------------------------
  const int cb = 256;
  k_f32_to_f16<<<(G4 * E + cb - 1) / cb, cb, 0, stream>>>(Wih_f, wihf16, G4 * E);
  k_f32_to_f16<<<(G4 * H + cb - 1) / cb, cb, 0, stream>>>(Whh_f, whhf16, G4 * H);
  k_f32_to_f16<<<(G4 * E + cb - 1) / cb, cb, 0, stream>>>(Wih_b, wihb16, G4 * E);
  k_f32_to_f16<<<(G4 * H + cb - 1) / cb, cb, 0, stream>>>(Whh_b, whhb16, G4 * H);
  k_f32_to_f16<<<(512 * 512 + cb - 1) / cb, cb, 0, stream>>>(Wimg, wimg16, 512 * 512);

  // ---- embedding gather ------------------------------------------------
  k_gather_embed<<<NT * L, 128, 0, stream>>>(tokens, embed, x16);

  // ---- bidirectional LSTM (fused input+recurrent GEMMs) ---------------
  k_lstm_dir<<<NT / 16, 128, 0, stream>>>(x16, wihf16, whhf16, bih_f, bhh_f, h_f, 0);
  k_lstm_dir<<<NT / 16, 128, 0, stream>>>(x16, wihb16, whhb16, bih_b, bhh_b, h_b, 1);

  // ---- image branch ----------------------------------------------------
  k_image_mean<<<NT, 128, 0, stream>>>(images, imean16);
  dim3 gg(NT / 16, 512 / 64);
  k_gemm_bias<<<gg, 128, 0, stream>>>(imean16, wimg16, bimg, image_h, NT, 512, 512);

  // ---- fusion + classifier --------------------------------------------
  k_fusion<<<NT, 128, 0, stream>>>(h_f, h_b, image_h, W_T, W_I, feature);
  k_col_mean<<<4, 128, 0, stream>>>(feature, mf);
  k_classifier<<<1, 256, 0, stream>>>(mf, Wc1, bc1, Wc2, bc2, out);

  (void)in_sizes; (void)n_in; (void)out_size; (void)ws_size;
}
